// LogSignatureHedge_48473000902871
// MI455X (gfx1250) — matrix operand order repaired and checked by reading, not compiled, not run on hardware
//
#include <hip/hip_runtime.h>
#include <hip/hip_bf16.h>

// ---------------------------------------------------------------------------
// LogSignatureHedge for MI455X (gfx1250, wave32, WMMA)
//
// Phase A (per batch row, 1024 blocks x 256 threads):
//   - stage features (256x10) in LDS
//   - branch-free logsig factorization: every static-x column j is
//       x[k][j] = U[k][idx1[j]] * U[k][idx2[j]]
//     with U[k] = [sig-one | rel[k] (10) | sig2f[k] (121, inv_k pre-scaled) |
//                  features[k] (10) | 0 | 1], built by a 121-channel
//     sequential cumsum pass + uniform fills.
//   - W1[0:522] f32->f16, swizzled into WMMA B-fragment layout in LDS
//   - 16x(16-row) tiles: A fragments assembled from U via packed index table
//     (pure ds_load + v_mul + v_cvt_pk, no divergence), 17 K-chunks,
//     v_wmma_f32_16x16x32_f16 seeded with b1; pre1 (B*N x 64 f32) -> d_ws.
//
// Phase B (64 blocks x 32 threads; 16 sequences per wave):
//   - W2 swizzled to f16 B-fragments in LDS
//   - 256 sequential steps: h1 = relu(pre1 + delta*W1[522]) as A frags,
//     two chained WMMAs per 16-col tile (K=64), relu, dot with W3 via a
//     wave32 __shfl_xor butterfly, write delta, feed it forward.
//
// Workspace: pre1 = 1024*256*64 f32 = 64 MB in d_ws.
// ---------------------------------------------------------------------------

typedef __attribute__((ext_vector_type(16))) _Float16 v16h;
typedef __attribute__((ext_vector_type(8)))  float    v8f;

#define BSZ   1024
#define NSZ   256
#define DIN   10
#define HID   64
#define LSD   512
#define KSTAT 522          // 512 logsig + 10 features (delta column excluded)
#define KPAD  544          // NCH * 32
#define NCH   17           // ceil(522/32) K-chunks of 32
#define USTR  145          // U-table row stride (floats, odd for bank spread)

#define U_ONEK   0         // 1 for k>0 else 0  (sig1[0] and sig3 factor r=0)
#define U_REL    1         // [1..10]  rel[k][i] = f[k][i]-f[0][i]
#define U_SIG2   11        // [11..131] sig2 flattened 11x11, inv_k pre-scaled
#define U_FEAT   132       // [132..141] raw features[k]
#define U_ZERO   142       // always 0
#define U_ONE    143       // always 1

__device__ __forceinline__ unsigned pack2h(float a, float b) {
  unsigned short ha = __builtin_bit_cast(unsigned short, (_Float16)a);
  unsigned short hb = __builtin_bit_cast(unsigned short, (_Float16)b);
  return (unsigned)ha | ((unsigned)hb << 16);
}

// ---------------------------------------------------------------------------
// Phase A
// ---------------------------------------------------------------------------
__global__ __launch_bounds__(256)
void logsig_gemm_kernel(const float* __restrict__ features,
                        const float* __restrict__ W1,
                        const float* __restrict__ b1,
                        float* __restrict__ pre1) {
  __shared__ float    ls_feat[NSZ * DIN];                    // 10.0 KB
  __shared__ unsigned ls_idx[KPAD];                          //  2.1 KB
  __shared__ float    ls_u[NSZ * USTR];                      // 145.0 KB
  __shared__ __align__(32) unsigned ls_w1[NCH * 4 * 32 * 8]; // 68.0 KB

  const int tid   = threadIdx.x;
  const int b     = blockIdx.x;
  const int lane  = tid & 31;
  const int wv    = tid >> 5;       // 8 waves
  const int mlane = lane & 15;
  const int g     = lane >> 4;

  // stage features for this batch row
  for (int i = tid; i < NSZ * DIN; i += 256)
    ls_feat[i] = features[(size_t)b * (NSZ * DIN) + i];

  // factor-index table: column j -> (idx1, idx2) into U[k]
  for (int j = tid; j < KPAD; j += 256) {
    unsigned i1, i2;
    if (j < 132)               { i1 = (unsigned)j;            i2 = U_ONE;  }
    else if (j < LSD)          { int q = j - 132;
                                 int r = q / 121, c = q - r * 121;
                                 i1 = (unsigned)r;            i2 = U_SIG2 + c; }
    else if (j < KSTAT)        { i1 = U_FEAT + (j - LSD);     i2 = U_ONE;  }
    else                       { i1 = U_ZERO;                 i2 = U_ZERO; }
    ls_idx[j] = i1 | (i2 << 16);
  }
  __syncthreads();   // features ready

  // zero U (covers the k==0 all-zero logsig row and unwritten pads)
  for (int i = tid; i < NSZ * USTR; i += 256) ls_u[i] = 0.0f;

  // W1 (static rows only) -> f16, swizzled to B-fragment layout:
// word idx = ((ch*4 + nt)*32 + l)*8 + v ; lane l holds col n = nt*16+(l&15),
// K = ch*32 + 8*(l>>4) + 2*(v&3) + 16*(v>>2) (+1 in hi half).
  for (int idx = tid; idx < NCH * 4 * 32 * 8; idx += 256) {
    int v  = idx & 7;
    int l  = (idx >> 3) & 31;
    int nt = (idx >> 8) & 3;
    int c  = idx >> 10;
    int k0 = c * 32 + 8 * (l >> 4) + 2 * (v & 3) + 16 * (v >> 2);
    int n  = nt * 16 + (l & 15);
    float f0 = (k0 < KSTAT)     ? W1[k0 * HID + n]       : 0.0f;
    float f1 = (k0 + 1 < KSTAT) ? W1[(k0 + 1) * HID + n] : 0.0f;
    ls_w1[idx] = pack2h(f0, f1);
  }
  __syncthreads();   // U zeroed

  // uniform per-row fills: rel, features, constants
  for (int idx = tid; idx < NSZ * 16; idx += 256) {
    int k = idx >> 4, s = idx & 15;
    if (s < DIN) {
      float f = ls_feat[k * DIN + s];
      ls_u[k * USTR + U_FEAT + s] = f;
      if (k > 0) ls_u[k * USTR + U_REL + s] = f - ls_feat[s];
    } else if (s == 10) {
      ls_u[k * USTR + U_ONEK] = (k > 0) ? 1.0f : 0.0f;
    } else if (s == 11) {
      ls_u[k * USTR + U_ONE] = 1.0f;
    }
  }

  // sequential cumulative sums, one channel per thread, scaled on store.
  // row k = t+1 uses cumsums through t and inv_k = 1/k.
  if (tid < 121) {
    float acc = 0.0f;
    if (tid < 10) {                       // inv_k * A_i,  sig2 (0, i+1)
      int i = tid;
      float prev = ls_feat[i];
      for (int t = 0; t < NSZ - 1; ++t) {
        float nxt = ls_feat[(t + 1) * DIN + i];
        acc += (float)t * (nxt - prev);
        ls_u[(t + 1) * USTR + U_SIG2 + (i + 1)] = acc / (float)(t + 1);
        prev = nxt;
      }
    } else if (tid < 20) {                // inv_k * Bc_i, sig2 (i+1, 0)
      int i = tid - 10;
      float f0 = ls_feat[i];
      for (int t = 0; t < NSZ - 1; ++t) {
        acc += ls_feat[t * DIN + i] - f0;
        ls_u[(t + 1) * USTR + U_SIG2 + (i + 1) * 11] = acc / (float)(t + 1);
      }
    } else if (tid < 120) {               // C_ij, sig2 (i+1, j+1)
      int p = tid - 20;
      int i = p / 10, j = p - i * 10;
      float f0i = ls_feat[i];
      float fj  = ls_feat[j];
      for (int t = 0; t < NSZ - 1; ++t) {
        float fi  = ls_feat[t * DIN + i];
        float fjn = ls_feat[(t + 1) * DIN + j];
        acc += (fi - f0i) * (fjn - fj);
        ls_u[(t + 1) * USTR + U_SIG2 + (i + 1) * 11 + (j + 1)] = acc;
        fj = fjn;
      }
    } else {                              // tt = 0.5*(k-1)/k, sig2 (0,0)
      for (int t = 0; t < NSZ - 1; ++t)
        ls_u[(t + 1) * USTR + U_SIG2] = 0.5f * (float)t / (float)(t + 1);
    }
  }
  __syncthreads();

  float b1v[4];
#pragma unroll
  for (int nt = 0; nt < 4; ++nt) b1v[nt] = b1[nt * 16 + mlane];

  // 16 row-tiles of 16 prefix rows each; 2 tiles per wave
#pragma unroll 1
  for (int tile = wv; tile < 16; tile += 8) {
    const int krow = tile * 16 + mlane;                  // A-frag row M
    const int ub   = krow * USTR;

    v8f acc[4];
#pragma unroll
    for (int nt = 0; nt < 4; ++nt)
      acc[nt] = (v8f){b1v[nt], b1v[nt], b1v[nt], b1v[nt],
                      b1v[nt], b1v[nt], b1v[nt], b1v[nt]};

#pragma unroll 1
    for (int ch = 0; ch < NCH; ++ch) {
      // 16-bit A 16x32: elems 0..7 -> K = 8g+0..7 ; 8..15 -> K = 16+8g+0..7
      v16h a;
      const int jb0 = ch * 32 + (g << 3);
#pragma unroll
      for (int i = 0; i < 8; ++i) {
        unsigned ix = ls_idx[jb0 + i];
        float val = ls_u[ub + (ix & 0xffffu)] * ls_u[ub + (ix >> 16)];
        a[i] = (_Float16)val;
      }
#pragma unroll
      for (int i = 0; i < 8; ++i) {
        unsigned ix = ls_idx[jb0 + 16 + i];
        float val = ls_u[ub + (ix & 0xffffu)] * ls_u[ub + (ix >> 16)];
        a[8 + i] = (_Float16)val;
      }

      const int fb = ch * 1024 + lane * 8;
#pragma unroll
      for (int nt = 0; nt < 4; ++nt) {
        v16h bf = *reinterpret_cast<const v16h*>(&ls_w1[fb + nt * 256]);
        acc[nt] = __builtin_amdgcn_wmma_f32_16x16x32_f16(
            false, a, false, bf, (short)0, acc[nt], false, false);
      }
    }

    // store pre1: D layout lane holds col n = nt*16+mlane, VGPR v -> m = v+8g
#pragma unroll
    for (int nt = 0; nt < 4; ++nt) {
      const int n = nt * 16 + mlane;
#pragma unroll
      for (int v = 0; v < 8; ++v) {
        int k = tile * 16 + v + 8 * g;
        pre1[((size_t)b * NSZ + k) * HID + n] = acc[nt][v];
      }
    }
  }
}

// ---------------------------------------------------------------------------
// Phase B: 16 sequences per wave, 256 sequential steps
// ---------------------------------------------------------------------------
__global__ __launch_bounds__(32)
void scan_kernel(const float* __restrict__ pre1,
                 const float* __restrict__ W1,
                 const float* __restrict__ W2,
                 const float* __restrict__ b2,
                 const float* __restrict__ W3,
                 const float* __restrict__ b3,
                 float* __restrict__ out) {
  __shared__ __align__(32) unsigned ls_w2[2 * 4 * 32 * 8];   // 8 KB

  const int lane  = threadIdx.x & 31;
  const int mlane = lane & 15;
  const int g     = lane >> 4;
  const int b0    = blockIdx.x * 16;

  // W2 -> f16 B fragments (2 K-chunks x 4 N-tiles)
  for (int idx = threadIdx.x; idx < 2048; idx += 32) {
    int v  = idx & 7;
    int l  = (idx >> 3) & 31;
    int nt = (idx >> 8) & 3;
    int c  = idx >> 10;
    int k0 = c * 32 + 8 * (l >> 4) + 2 * (v & 3) + 16 * (v >> 2);
    int n  = nt * 16 + (l & 15);
    ls_w2[idx] = pack2h(W2[k0 * HID + n], W2[(k0 + 1) * HID + n]);
  }
  __syncthreads();

  float w3v[4], b2v[4];
#pragma unroll
  for (int nt = 0; nt < 4; ++nt) {
    w3v[nt] = W3[nt * 16 + mlane];
    b2v[nt] = b2[nt * 16 + mlane];
  }
  const float b3s = b3[0];

  // delta column of W1, in the per-lane A-fragment slot order
  float w1l[2][16];
#pragma unroll
  for (int ch = 0; ch < 2; ++ch) {
    int h0 = ch * 32 + 8 * g, h1 = h0 + 16;
#pragma unroll
    for (int i = 0; i < 8; ++i) {
      w1l[ch][i]     = W1[KSTAT * HID + h0 + i];
      w1l[ch][8 + i] = W1[KSTAT * HID + h1 + i];
    }
  }

  const size_t bm = (size_t)(b0 + mlane);   // this lane's sequence
  float dmine = 0.0f;                       // prev-step delta for row mlane

#pragma unroll 1
  for (int t = 0; t < NSZ; ++t) {
    const float* row = pre1 + (bm * NSZ + t) * HID;

    v16h a[2];
#pragma unroll
    for (int ch = 0; ch < 2; ++ch) {
      int h0 = ch * 32 + 8 * g, h1 = h0 + 16;
#pragma unroll
      for (int i = 0; i < 8; ++i) {
        float x0 = fmaxf(row[h0 + i] + dmine * w1l[ch][i],     0.0f);
        float x1 = fmaxf(row[h1 + i] + dmine * w1l[ch][8 + i], 0.0f);
        a[ch][i]     = (_Float16)x0;
        a[ch][8 + i] = (_Float16)x1;
      }
    }

    v8f acc[4];
#pragma unroll
    for (int nt = 0; nt < 4; ++nt) {
      acc[nt] = (v8f){b2v[nt], b2v[nt], b2v[nt], b2v[nt],
                      b2v[nt], b2v[nt], b2v[nt], b2v[nt]};
      v16h bf0 = *reinterpret_cast<const v16h*>(&ls_w2[nt * 256 + lane * 8]);
      acc[nt] = __builtin_amdgcn_wmma_f32_16x16x32_f16(
          false, a[0], false, bf0, (short)0, acc[nt], false, false);
      v16h bf1 = *reinterpret_cast<const v16h*>(&ls_w2[1024 + nt * 256 + lane * 8]);
      acc[nt] = __builtin_amdgcn_wmma_f32_16x16x32_f16(
          false, a[1], false, bf1, (short)0, acc[nt], false, false);
    }

    // delta[m] = sum_n relu(h2[m,n]) * W3[n] + b3 ; reduce across lanes
    float p[8];
#pragma unroll
    for (int v = 0; v < 8; ++v) p[v] = 0.0f;
#pragma unroll
    for (int nt = 0; nt < 4; ++nt)
#pragma unroll
      for (int v = 0; v < 8; ++v)
        p[v] += fmaxf(acc[nt][v], 0.0f) * w3v[nt];
#pragma unroll
    for (int v = 0; v < 8; ++v) {
      p[v] += __shfl_xor(p[v], 1, 32);
      p[v] += __shfl_xor(p[v], 2, 32);
      p[v] += __shfl_xor(p[v], 4, 32);
      p[v] += __shfl_xor(p[v], 8, 32);      // now p[v] = delta for m = v + 8g
    }
    float q[8];
#pragma unroll
    for (int v = 0; v < 8; ++v) q[v] = __shfl_xor(p[v], 16, 32); // m = v+8(1-g)

    const int hi = mlane >> 3, vi = mlane & 7;
    float r = 0.0f;
#pragma unroll
    for (int v = 0; v < 8; ++v)
      if (vi == v) r = (hi == g) ? p[v] : q[v];

    const float dnew = r + b3s;
    if (g == 0) out[bm * NSZ + t] = dnew;   // one writer per sequence
    dmine = dnew;                           // feed forward
  }
}

// ---------------------------------------------------------------------------
extern "C" void kernel_launch(void* const* d_in, const int* in_sizes, int n_in,
                              void* d_out, int out_size, void* d_ws, size_t ws_size,
                              hipStream_t stream) {
  const float* features = (const float*)d_in[0];   // (1024, 256, 10)
  const float* W1       = (const float*)d_in[1];   // (523, 64)
  const float* b1       = (const float*)d_in[2];   // (64,)
  const float* W2       = (const float*)d_in[3];   // (64, 64)
  const float* b2       = (const float*)d_in[4];   // (64,)
  const float* W3       = (const float*)d_in[5];   // (64, 1)
  const float* b3       = (const float*)d_in[6];   // (1,)
  float* out  = (float*)d_out;                     // (1024, 256)
  float* pre1 = (float*)d_ws;                      // 1024*256*64 f32 = 64 MB

  logsig_gemm_kernel<<<BSZ, 256, 0, stream>>>(features, W1, b1, pre1);
  scan_kernel<<<BSZ / 16, 32, 0, stream>>>(pre1, W1, W2, b2, W3, b3, out);
}
